// AWQW4A16Linear_24627342475387
// MI455X (gfx1250) — matrix operand order repaired and checked by reading, not compiled, not run on hardware
//
#include <hip/hip_runtime.h>

// ---------------------------------------------------------------------------
// AWQ W4A16 GEMM for MI455X (gfx1250, wave32, WMMA bf16 16x16x32)
//   out[M,N] = x[M,K] @ dequant(qweight)[K,N] + bias
//   M=8192, K=4096, N=14336, GROUP=64
// Compute-bound (~20x over the 23.3 TB/s HBM roofline):
//   - bf16 WMMA, fused int4 dequant into LDS (packed bf16 ALU, no f32 round trip)
//   - double-buffered LDS tiles, one barrier per K-iteration
//   - x tile staged with GLOBAL_LOAD_ASYNC_TO_LDS_B128 (ASYNCcnt) issued after
//     the barrier so next-tile traffic hides under current-tile WMMA
//   - qweight/scale/zero for the next tile register-staged after the barrier;
//     dequant VALU co-executes with the WMMA (TRANS) pipe
// Dequant trick: 0x4300|nib is bf16(128+nib) exactly, so
//   (w-z)*s == ((128+w)-(128+z))*s  with an exact bf16 subtract and a single
//   RNE multiply -> bit-identical to the reference bf16 math.
// ---------------------------------------------------------------------------

typedef __attribute__((ext_vector_type(16))) __bf16 v16bf;
typedef __attribute__((ext_vector_type(2)))  __bf16 v2bf;
typedef __attribute__((ext_vector_type(8)))  float  v8f;
typedef int v4i_vs __attribute__((vector_size(16)));   // matches builtin proto

constexpr int K  = 4096;
constexpr int N  = 14336;

constexpr int BM = 128;      // M tile per block
constexpr int BN = 128;      // N tile per block
constexpr int BK = 64;       // K tile (== GROUP: one scale/zero per tile)
constexpr int LDA = BK + 8;  // padded LDS stride (halves): 144B, 16B-aligned
constexpr int LDB = BK + 8;
constexpr int KT = K / BK;   // 64 iterations

#if __has_builtin(__builtin_amdgcn_global_load_async_to_lds_b128)
#define USE_ASYNC_LDS 1
#else
#define USE_ASYNC_LDS 0
#endif

// 16-byte global -> LDS copy. LDS address passed as the 32-bit AS(3) offset
// (low 32 bits of the generic pointer are the wave-relative LDS offset).
__device__ __forceinline__ void async_copy16(const unsigned short* g,
                                             unsigned short* l) {
#if USE_ASYNC_LDS
    __builtin_amdgcn_global_load_async_to_lds_b128(
        (__attribute__((address_space(1))) v4i_vs*)(unsigned long long)g,
        (__attribute__((address_space(3))) v4i_vs*)(unsigned int)(unsigned long long)l,
        0, 0);
#else
    *(uint4*)l = *(const uint4*)g;
#endif
}

__device__ __forceinline__ void wait_async0() {
#if USE_ASYNC_LDS
#if __has_builtin(__builtin_amdgcn_s_wait_asynccnt)
    __builtin_amdgcn_s_wait_asynccnt(0);
#else
    asm volatile("s_wait_asynccnt 0x0" ::: "memory");
#endif
#endif
}

__device__ __forceinline__ float bf16_to_f32(unsigned short h) {
    unsigned int u = (unsigned int)h << 16;
    float f;
    __builtin_memcpy(&f, &u, 4);
    return f;
}

__global__ __launch_bounds__(256)
void awq_w4a16_wmma(const unsigned short* __restrict__ x,       // bf16 [M,K]
                    const unsigned int*   __restrict__ qweight, // i32  [K/8,N]
                    const unsigned short* __restrict__ scales,  // bf16 [K/64,N]
                    const unsigned int*   __restrict__ qzeros,  // i32  [K/64,N/8]
                    const unsigned short* __restrict__ bias,    // bf16 [N]
                    unsigned short*       __restrict__ out)     // bf16 [M,N]
{
    __shared__ unsigned short sA[2][BM * LDA];  // x tiles,  [BM][BK]
    __shared__ unsigned short sB[2][BN * LDB];  // w tiles,  [BN][BK] (K-contig)

    const int tid = threadIdx.x;
    const int bn0 = blockIdx.x * BN;
    const int bm0 = blockIdx.y * BM;

    const int wave  = tid >> 5;
    const int lane  = tid & 31;
    const int wm    = wave >> 2;    // 0..1 : 64 rows each
    const int wn    = wave & 3;     // 0..3 : 32 cols each
    const int lrow  = lane & 15;
    const int lhalf = lane >> 4;

    // dequant assignment: thread owns one tile column, 4 packed-K rows
    const int dn   = tid & 127;
    const int dr   = tid >> 7;      // r8 = dr + 2*i
    const int gcol = bn0 + dn;

    v8f acc[4][2];
    {
        v8f z = {};
        #pragma unroll
        for (int mi = 0; mi < 4; ++mi)
            #pragma unroll
            for (int ni = 0; ni < 2; ++ni) acc[mi][ni] = z;
    }

    unsigned int qv[4];     // next tile's packed weights (register-staged)
    unsigned int zbits;     // bf16(128+z) splat in both halves
    unsigned int sbits;     // scale bf16 splat in both halves

    auto stage_a = [&](int kt, int buf) {
        #pragma unroll
        for (int i = 0; i < 4; ++i) {
            int c   = tid + i * 256;           // 1024 chunks of 8 halves
            int row = c >> 3;
            int col = (c & 7) * 8;
            async_copy16(x + (size_t)(bm0 + row) * K + kt * BK + col,
                         &sA[buf][row * LDA + col]);
        }
    };
    auto load_q = [&](int kt) {
        unsigned int qz = qzeros[(size_t)kt * (N / 8) + (gcol >> 3)];
        unsigned int z4 = (qz >> (4 * (dn & 7))) & 15u;
        zbits = 0x43004300u | z4 | (z4 << 16);
        unsigned int s16 = scales[(size_t)kt * N + gcol];
        sbits = s16 | (s16 << 16);
        #pragma unroll
        for (int i = 0; i < 4; ++i)
            qv[i] = qweight[(size_t)(kt * 8 + (dr + 2 * i)) * N + gcol];
    };

    // prologue: kick off tile 0
    stage_a(0, 0);
    load_q(0);

    for (int kt = 0; kt < KT; ++kt) {
        const int cur = kt & 1;

        // ---- dequant registers (loaded last iteration) -> sB[cur] ----
        {
            const v2bf zv = __builtin_bit_cast(v2bf, zbits);
            const v2bf sv = __builtin_bit_cast(v2bf, sbits);
            #pragma unroll
            for (int i = 0; i < 4; ++i) {
                int r8 = dr + 2 * i;
                union { uint4 u; unsigned int pr[4]; } w;
                #pragma unroll
                for (int p = 0; p < 4; ++p) {       // nibble pair (2p, 2p+1)
                    unsigned int t = qv[i] >> (8 * p);
                    unsigned int bits = 0x43004300u | (t & 0xFu)
                                        | ((t & 0xF0u) << 12);
                    v2bf wv = __builtin_bit_cast(v2bf, bits);
                    v2bf r  = (wv - zv) * sv;        // v_pk_add/mul_bf16
                    w.pr[p] = __builtin_bit_cast(unsigned int, r);
                }
                *(uint4*)(&sB[cur][dn * LDB + r8 * 8]) = w.u;
            }
        }

        wait_async0();       // sA[cur] fully landed in LDS
        __syncthreads();     // cur writes visible; all reads of nxt finished

        if (kt + 1 < KT) {   // next tile's global traffic under this compute
            stage_a(kt + 1, cur ^ 1);
            load_q(kt + 1);
        }

        // ---- compute: 2 k-steps of 32, 8 WMMA tiles per wave per step ----
        #pragma unroll
        for (int k0 = 0; k0 < BK; k0 += 32) {
            union Frag { v16bf v; uint4 u[2]; };
            Frag a[4], b[2];

            // A 16x32 bf16: lanes 0-15 K {k0..+7, k0+16..+23}; lanes 16-31 +8
            #pragma unroll
            for (int mi = 0; mi < 4; ++mi) {
                int m  = wm * 64 + mi * 16 + lrow;
                int kk = k0 + lhalf * 8;
                a[mi].u[0] = *(const uint4*)(&sA[cur][m * LDA + kk]);
                a[mi].u[1] = *(const uint4*)(&sA[cur][m * LDA + kk + 16]);
            }
            // B 32x16 bf16: lane = column; lanes 0-15 K {k0..+7, k0+16..+23}; lanes 16-31 +8
            #pragma unroll
            for (int ni = 0; ni < 2; ++ni) {
                int n  = wn * 32 + ni * 16 + lrow;
                int kb = k0 + lhalf * 8;
                b[ni].u[0] = *(const uint4*)(&sB[cur][n * LDB + kb]);
                b[ni].u[1] = *(const uint4*)(&sB[cur][n * LDB + kb + 16]);
            }

            #pragma unroll
            for (int mi = 0; mi < 4; ++mi)
                #pragma unroll
                for (int ni = 0; ni < 2; ++ni)
                    acc[mi][ni] = __builtin_amdgcn_wmma_f32_16x16x32_bf16(
                        /*neg_a=*/false, a[mi].v,
                        /*neg_b=*/false, b[ni].v,
                        /*c_mod=*/(short)0, acc[mi][ni],
                        /*reuse_a=*/false, /*reuse_b=*/false);
        }
        // no trailing barrier: next iteration's barrier fences the buffers
    }

    // ---- epilogue: D layout -> VGPR vi: M = vi + 8*lhalf, N = lrow ----
    #pragma unroll
    for (int ni = 0; ni < 2; ++ni) {
        int gn   = bn0 + wn * 32 + ni * 16 + lrow;
        float bv = bf16_to_f32(bias[gn]);
        #pragma unroll
        for (int mi = 0; mi < 4; ++mi) {
            #pragma unroll
            for (int vi = 0; vi < 8; ++vi) {
                int gm = bm0 + wm * 64 + mi * 16 + lhalf * 8 + vi;
                __bf16 r = (__bf16)(acc[mi][ni][vi] + bv);   // v_cvt RNE
                out[(size_t)gm * N + gn] = __builtin_bit_cast(unsigned short, r);
            }
        }
    }
}

extern "C" void kernel_launch(void* const* d_in, const int* in_sizes, int n_in,
                              void* d_out, int out_size, void* d_ws, size_t ws_size,
                              hipStream_t stream) {
    const unsigned short* x       = (const unsigned short*)d_in[0]; // bf16
    const unsigned int*   qweight = (const unsigned int*)  d_in[1]; // int32
    const unsigned short* scales  = (const unsigned short*)d_in[2]; // bf16
    const unsigned int*   qzeros  = (const unsigned int*)  d_in[3]; // int32
    const unsigned short* bias    = (const unsigned short*)d_in[4]; // bf16
    unsigned short*       out     = (unsigned short*)d_out;         // bf16

    const int M = in_sizes[0] / K;               // 8192
    dim3 grid(N / BN, M / BM);                   // 112 x 64
    awq_w4a16_wmma<<<grid, dim3(256), 0, stream>>>(x, qweight, scales, qzeros,
                                                   bias, out);
}
